// MarketDynamicsEngine_37752762532576
// MI455X (gfx1250) — matrix operand order repaired and checked by reading, not compiled, not run on hardware
//
#include <hip/hip_runtime.h>
#include <hip/hip_bf16.h>

typedef unsigned short u16;
typedef unsigned int   u32;
typedef __attribute__((ext_vector_type(16))) __bf16 v16bf;
typedef __attribute__((ext_vector_type(8)))  float  v8f;
typedef __attribute__((ext_vector_type(4)))  u32    u32x4;
typedef __attribute__((ext_vector_type(2)))  u32    u32x2;
typedef __attribute__((ext_vector_type(4)))  float  f32x4;

union FragBF { v16bf v; u32x4 q[2]; };

__device__ __forceinline__ u16 f2bf(float f) {
  u32 u = __float_as_uint(f);
  u32 r = u + 0x7FFFu + ((u >> 16) & 1u);   // round-to-nearest-even
  return (u16)(r >> 16);
}
__device__ __forceinline__ float sigmoidf_(float x) { return 1.f / (1.f + expf(-x)); }

// ---- CDNA5 async global->LDS copy (ASYNCcnt tracked), 16B per lane ----
__device__ __forceinline__ void async_ld_b128(u32 lds_byte_off, const void* gaddr) {
  asm volatile("global_load_async_to_lds_b128 %0, %1, off"
               :: "v"(lds_byte_off), "v"(gaddr) : "memory");
}
__device__ __forceinline__ void wait_async0() {
  asm volatile("s_wait_asynccnt 0x0" ::: "memory");
}

// ---------------- problem constants ----------------
constexpr int NC = 16384, DIN = 512, DH = 1024, DOUT = 512, MW = 128, H3 = 3072;

// ---------------- init: zero accumulators ----------------
__global__ void init_kernel(float* tacc, float* cvec, int* cnts) {
  int idx = blockIdx.x * 256 + threadIdx.x;          // grid 64 -> 16384
  if (idx < NC)  tacc[idx] = 0.f;
  if (idx < DOUT) cvec[idx] = 0.f;
  if (idx < 32)  cnts[idx] = 0;
}

// ---------------- f32 -> bf16 bulk convert (4 elems/thread) ----------------
__global__ void conv_bf16_kernel(u16* __restrict__ dst, const float* __restrict__ src) {
  int i = blockIdx.x * 256 + threadIdx.x;            // grid = total/1024
  f32x4 f = reinterpret_cast<const f32x4*>(src)[i];
  u16 h0 = f2bf(f.x), h1 = f2bf(f.y), h2 = f2bf(f.z), h3 = f2bf(f.w);
  u32x2 o; o.x = (u32)h0 | ((u32)h1 << 16); o.y = (u32)h2 | ((u32)h3 << 16);
  reinterpret_cast<u32x2*>(dst)[i] = o;
}

// ---------------- transpose + convert: dst[n][k] = scale * src[k][n] ----------------
__global__ void transpose_conv_kernel(u16* __restrict__ dst, int dstStride,
                                      const float* __restrict__ src, int srcStride,
                                      int rowsK, int colsN, float scale) {
  int idx = blockIdx.x * 256 + threadIdx.x;
  if (idx >= rowsK * colsN) return;
  int n = idx / rowsK;
  int k = idx - n * rowsK;
  dst[(size_t)n * dstStride + k] = f2bf(scale * src[(size_t)k * srcStride + n]);
}

// ---------------- xa/xg bias: fold x @ W1[:512] + b1 into per-column bias ----------------
__global__ void xbias_kernel(const float* __restrict__ x,
                             const float* __restrict__ Wa1, const float* __restrict__ ba1,
                             const float* __restrict__ Wg1, const float* __restrict__ bg1,
                             float* __restrict__ xabias) {
  int n = threadIdx.x;                               // 1 block, 256 threads
  if (n < MW) {
    float s = ba1[n];
    for (int k = 0; k < DIN; ++k) s += x[k] * Wa1[(size_t)k * MW + n];
    xabias[n] = s;
  } else {
    int m = n - MW;
    float s = bg1[m];
    for (int k = 0; k < DIN; ++k) s += x[k] * Wg1[(size_t)k * MW + m];
    xabias[n] = s;
  }
}

// ---------------- generic bf16 WMMA GEMM, WG tile 128x128, async double-buffered ----------------
// A: [M][K] bf16 row-major.  B: [N][K] bf16 (K-major, i.e. W^T).
// EPI 0: out = relu(acc + bias1[col])        -> bf16 store (layer-1)
// EPI 1: out = acc + bias1[col] - bias2[col] -> f32 + bf16 stores + tension sum-of-squares
template <int EPI, int K>
__global__ __launch_bounds__(256)
void gemm_bf16_kernel(const u16* __restrict__ A, const u16* __restrict__ B,
                      int ldc,
                      u16* __restrict__ outBf, float* __restrict__ outF,
                      const float* __restrict__ bias1, const float* __restrict__ bias2,
                      float* __restrict__ tacc) {
  __shared__ __align__(16) u16 ldsA[2][128 * 40];
  __shared__ __align__(16) u16 ldsB[2][128 * 40];
  const int tid   = threadIdx.x;
  const int lane  = tid & 31;
  const int wave  = tid >> 5;
  const int waveM = wave & 3;
  const int waveN = wave >> 2;
  const int nl    = lane & 15;
  const int g     = lane >> 4;
  const int mBase = blockIdx.x * 128;
  const int nBase = blockIdx.y * 128;

  const u32 aLds = (u32)(uintptr_t)&ldsA[0][0];
  const u32 bLds = (u32)(uintptr_t)&ldsB[0][0];

  const v8f vzero = {0.f,0.f,0.f,0.f,0.f,0.f,0.f,0.f};
  v8f acc[2][4];
#pragma unroll
  for (int mi = 0; mi < 2; ++mi)
#pragma unroll
    for (int ni = 0; ni < 4; ++ni) acc[mi][ni] = vzero;

  auto issue = [&](int buf, int k0) {
#pragma unroll
    for (int i = 0; i < 2; ++i) {
      int chunk = tid + i * 256;                     // 512 chunks of 8 halfs per tile
      int row = chunk >> 2;
      int kc  = (chunk & 3) * 8;
      u32 lo = (u32)(buf * (128 * 40) + row * 40 + kc) * 2;
      async_ld_b128(aLds + lo, &A[(size_t)(mBase + row) * K + k0 + kc]);
      async_ld_b128(bLds + lo, &B[(size_t)(nBase + row) * K + k0 + kc]);
    }
  };

  constexpr int nk = K / 32;
  issue(0, 0);
#pragma unroll
  for (int it = 0; it < nk; ++it) {
    const int cur = it & 1;
    wait_async0();            // my current-buffer loads have landed in LDS
    __syncthreads();          // everyone's have; prev compute on other buffer done
    if (it + 1 < nk) issue(cur ^ 1, (it + 1) * 32);

    FragBF a[2], b[4];
#pragma unroll
    for (int mi = 0; mi < 2; ++mi) {
      int m = waveM * 32 + mi * 16 + nl;
      a[mi].q[0] = *reinterpret_cast<const u32x4*>(&ldsA[cur][m * 40 + 8 * g]);
      a[mi].q[1] = *reinterpret_cast<const u32x4*>(&ldsA[cur][m * 40 + 16 + 8 * g]);
    }
#pragma unroll
    for (int ni = 0; ni < 4; ++ni) {
      int n = waveN * 64 + ni * 16 + nl;
      b[ni].q[0] = *reinterpret_cast<const u32x4*>(&ldsB[cur][n * 40 + 8 * g]);
      b[ni].q[1] = *reinterpret_cast<const u32x4*>(&ldsB[cur][n * 40 + 16 + 8 * g]);
    }
#pragma unroll
    for (int mi = 0; mi < 2; ++mi)
#pragma unroll
      for (int ni = 0; ni < 4; ++ni)
        acc[mi][ni] = __builtin_amdgcn_wmma_f32_16x16x32_bf16(
            false, a[mi].v, false, b[ni].v, (short)0, acc[mi][ni], false, false);
  }

#pragma unroll
  for (int mi = 0; mi < 2; ++mi) {
#pragma unroll
    for (int v = 0; v < 8; ++v) {
      int row = mBase + waveM * 32 + mi * 16 + 8 * g + v;
      if constexpr (EPI == 0) {
#pragma unroll
        for (int ni = 0; ni < 4; ++ni) {
          int col = nBase + waveN * 64 + ni * 16 + nl;
          float o = acc[mi][ni][v] + bias1[col];
          o = fmaxf(o, 0.f);
          outBf[(size_t)row * ldc + col] = f2bf(o);
        }
      } else {
        float s = 0.f;
#pragma unroll
        for (int ni = 0; ni < 4; ++ni) {
          int col = nBase + waveN * 64 + ni * 16 + nl;
          float o = acc[mi][ni][v] + bias1[col] - bias2[col];
          outF[(size_t)row * ldc + col] = o;
          outBf[(size_t)row * ldc + col] = f2bf(o);
          s += o * o;
        }
        s += __shfl_xor(s, 1); s += __shfl_xor(s, 2);
        s += __shfl_xor(s, 4); s += __shfl_xor(s, 8);
        if (nl == 0) atomicAdd(&tacc[row], s);        // row sum over 64 cols
      }
    }
  }
}

// ---------------- tension finalize + softmax stats + avg tension ----------------
__global__ void tension_softmax_kernel(const float* __restrict__ tacc,
                                       float* __restrict__ tension,
                                       float* __restrict__ scal, float* __restrict__ avg_out) {
  __shared__ float red[256];
  int t = threadIdx.x;
  float lmax = -1e30f, lsum = 0.f;
  for (int i = t; i < NC; i += 256) {
    float x = tacc[i] * (1.f / (float)DOUT);
    tension[i] = x;
    lmax = fmaxf(lmax, x);
    lsum += x;
  }
  red[t] = lmax; __syncthreads();
  for (int s = 128; s > 0; s >>= 1) { if (t < s) red[t] = fmaxf(red[t], red[t + s]); __syncthreads(); }
  float m = red[0]; __syncthreads();
  red[t] = lsum; __syncthreads();
  for (int s = 128; s > 0; s >>= 1) { if (t < s) red[t] += red[t + s]; __syncthreads(); }
  float tsum = red[0]; __syncthreads();
  float lexp = 0.f;
  for (int i = t; i < NC; i += 256) lexp += expf(tension[i] - m);
  red[t] = lexp; __syncthreads();
  for (int s = 128; s > 0; s >>= 1) { if (t < s) red[t] += red[t + s]; __syncthreads(); }
  if (t == 0) { scal[0] = m; scal[1] = red[0]; avg_out[0] = tsum * (1.f / (float)NC); }
}

// ---------------- fused GRU: WG tile 64(M) x 64(hidden-col), async double-buffered ----------------
// Two compile-time phases with fixed accumulator sets (no runtime accumulator selection).
__global__ __launch_bounds__(256)
void gru_kernel(const u16* __restrict__ outBf,   // [NC][512] bf16 (output = a-g)
                const u16* __restrict__ hBf,     // [NC][1024] bf16
                const u16* __restrict__ WihT,    // [3072][512] bf16
                const u16* __restrict__ WhhT,    // [3072][1024] bf16
                const float* __restrict__ bih, const float* __restrict__ bhh,
                const float* __restrict__ wih512,      // W_ih row 512 (tension row), f32 [3072]
                const float* __restrict__ tension,
                const float* __restrict__ hiddens,     // original f32 h
                const float* __restrict__ wealth,
                float* __restrict__ newh) {
  __shared__ __align__(16) u16 lA[2][64 * 40];
  __shared__ __align__(16) u16 lB[2][3][64 * 40];
  const int tid   = threadIdx.x;
  const int lane  = tid & 31;
  const int wave  = tid >> 5;
  const int waveM = wave & 3;
  const int waveN = wave >> 2;
  const int nl    = lane & 15;
  const int g     = lane >> 4;
  const int mBase = blockIdx.x * 64;
  const int cBase = blockIdx.y * 64;
  const int lrow  = tid >> 2;
  const int lkc   = (tid & 3) * 8;

  const u32 aLds = (u32)(uintptr_t)&lA[0][0];
  const u32 bLds = (u32)(uintptr_t)&lB[0][0][0];

  const v8f vzero = {0.f,0.f,0.f,0.f,0.f,0.f,0.f,0.f};
  v8f aR[2], aZ[2], aN1[2], aN2[2];
#pragma unroll
  for (int ni = 0; ni < 2; ++ni) { aR[ni] = vzero; aZ[ni] = vzero; aN1[ni] = vzero; aN2[ni] = vzero; }

  auto issueP1 = [&](int buf, int k0) {              // mem_input tile + W_ih tiles
    async_ld_b128(aLds + (u32)(buf * (64 * 40) + lrow * 40 + lkc) * 2,
                  &outBf[(size_t)(mBase + lrow) * DIN + k0 + lkc]);
#pragma unroll
    for (int s = 0; s < 3; ++s)
      async_ld_b128(bLds + (u32)((buf * 3 + s) * (64 * 40) + lrow * 40 + lkc) * 2,
                    &WihT[(size_t)(s * DH + cBase + lrow) * DIN + k0 + lkc]);
  };
  auto issueP2 = [&](int buf, int k0) {              // h tile + W_hh tiles
    async_ld_b128(aLds + (u32)(buf * (64 * 40) + lrow * 40 + lkc) * 2,
                  &hBf[(size_t)(mBase + lrow) * DH + k0 + lkc]);
#pragma unroll
    for (int s = 0; s < 3; ++s)
      async_ld_b128(bLds + (u32)((buf * 3 + s) * (64 * 40) + lrow * 40 + lkc) * 2,
                    &WhhT[(size_t)(s * DH + cBase + lrow) * DH + k0 + lkc]);
  };

  constexpr int NK1 = DIN / 32;             // 16
  constexpr int NK2 = DH / 32;              // 32

  // ---- phase 1: K = 512 over mem_input (output bf16), gates r,z,n_in ----
  issueP1(0, 0);
#pragma unroll
  for (int it = 0; it < NK1; ++it) {
    const int cur = it & 1;
    wait_async0();
    __syncthreads();
    if (it + 1 < NK1) issueP1(cur ^ 1, (it + 1) * 32);
    else              issueP2(cur ^ 1, 0);           // prefetch phase-2 tile 0 (buf 0)

    FragBF a;
    {
      int m = waveM * 16 + nl;
      a.q[0] = *reinterpret_cast<const u32x4*>(&lA[cur][m * 40 + 8 * g]);
      a.q[1] = *reinterpret_cast<const u32x4*>(&lA[cur][m * 40 + 16 + 8 * g]);
    }
#pragma unroll
    for (int ni = 0; ni < 2; ++ni) {
      int n = waveN * 32 + ni * 16 + nl;
      FragBF br, bz, bn;
      br.q[0] = *reinterpret_cast<const u32x4*>(&lB[cur][0][n * 40 + 8 * g]);
      br.q[1] = *reinterpret_cast<const u32x4*>(&lB[cur][0][n * 40 + 16 + 8 * g]);
      bz.q[0] = *reinterpret_cast<const u32x4*>(&lB[cur][1][n * 40 + 8 * g]);
      bz.q[1] = *reinterpret_cast<const u32x4*>(&lB[cur][1][n * 40 + 16 + 8 * g]);
      bn.q[0] = *reinterpret_cast<const u32x4*>(&lB[cur][2][n * 40 + 8 * g]);
      bn.q[1] = *reinterpret_cast<const u32x4*>(&lB[cur][2][n * 40 + 16 + 8 * g]);
      aR[ni]  = __builtin_amdgcn_wmma_f32_16x16x32_bf16(false, a.v, false, br.v, (short)0, aR[ni],  false, false);
      aZ[ni]  = __builtin_amdgcn_wmma_f32_16x16x32_bf16(false, a.v, false, bz.v, (short)0, aZ[ni],  false, false);
      aN1[ni] = __builtin_amdgcn_wmma_f32_16x16x32_bf16(false, a.v, false, bn.v, (short)0, aN1[ni], false, false);
    }
  }

  // ---- phase 2: K = 1024 over h (bf16); r,z accumulate on; n_h separate ----
  // buffer parity continues: phase-2 tile 0 was prefetched into buf 0
#pragma unroll
  for (int it = 0; it < NK2; ++it) {
    const int cur = it & 1;
    wait_async0();
    __syncthreads();
    if (it + 1 < NK2) issueP2(cur ^ 1, (it + 1) * 32);

    FragBF a;
    {
      int m = waveM * 16 + nl;
      a.q[0] = *reinterpret_cast<const u32x4*>(&lA[cur][m * 40 + 8 * g]);
      a.q[1] = *reinterpret_cast<const u32x4*>(&lA[cur][m * 40 + 16 + 8 * g]);
    }
#pragma unroll
    for (int ni = 0; ni < 2; ++ni) {
      int n = waveN * 32 + ni * 16 + nl;
      FragBF br, bz, bn;
      br.q[0] = *reinterpret_cast<const u32x4*>(&lB[cur][0][n * 40 + 8 * g]);
      br.q[1] = *reinterpret_cast<const u32x4*>(&lB[cur][0][n * 40 + 16 + 8 * g]);
      bz.q[0] = *reinterpret_cast<const u32x4*>(&lB[cur][1][n * 40 + 8 * g]);
      bz.q[1] = *reinterpret_cast<const u32x4*>(&lB[cur][1][n * 40 + 16 + 8 * g]);
      bn.q[0] = *reinterpret_cast<const u32x4*>(&lB[cur][2][n * 40 + 8 * g]);
      bn.q[1] = *reinterpret_cast<const u32x4*>(&lB[cur][2][n * 40 + 16 + 8 * g]);
      aR[ni]  = __builtin_amdgcn_wmma_f32_16x16x32_bf16(false, a.v, false, br.v, (short)0, aR[ni],  false, false);
      aZ[ni]  = __builtin_amdgcn_wmma_f32_16x16x32_bf16(false, a.v, false, bz.v, (short)0, aZ[ni],  false, false);
      aN2[ni] = __builtin_amdgcn_wmma_f32_16x16x32_bf16(false, a.v, false, bn.v, (short)0, aN2[ni], false, false);
    }
  }

  // ---- epilogue: full GRU gate math + wealth mod + clip ----
#pragma unroll
  for (int ni = 0; ni < 2; ++ni) {
    int j = cBase + waveN * 32 + ni * 16 + nl;
    float br = bih[j] + bhh[j];
    float bz = bih[DH + j] + bhh[DH + j];
    float bni = bih[2 * DH + j];
    float bnh = bhh[2 * DH + j];
    float wr = wih512[j], wz = wih512[DH + j], wn = wih512[2 * DH + j];
#pragma unroll
    for (int v = 0; v < 8; ++v) {
      int row = mBase + waveM * 16 + 8 * g + v;
      float tn = tension[row];
      float r  = sigmoidf_(aR[ni][v] + br + tn * wr);
      float z  = sigmoidf_(aZ[ni][v] + bz + tn * wz);
      float ng = tanhf(aN1[ni][v] + bni + tn * wn + r * (aN2[ni][v] + bnh));
      float hold = hiddens[(size_t)row * DH + j];
      float hn = (1.f - z) * ng + z * hold;
      float wl = fminf(fmaxf(wealth[row], 0.1f), 2.0f);
      hn = hn * (0.9f + 0.1f * wl);
      hn = fminf(fmaxf(hn, -10.f), 10.f);
      newh[(size_t)row * DH + j] = hn;
    }
  }
}

// ---------------- sync stage 1a: mask / faction counts ----------------
__global__ void counts_kernel(const int* __restrict__ positions, int* __restrict__ cnts) {
  int i = blockIdx.x * 256 + threadIdx.x;            // grid 64
  int p = positions[i];
  int f = i >> 11;                                   // 2048 rows per faction
  if (p > 0)      { atomicAdd(&cnts[0], 1); atomicAdd(&cnts[2 + f], 1); }
  else if (p < 0) { atomicAdd(&cnts[1], 1); atomicAdd(&cnts[10 + f], 1); }
}

// ---------------- sync stage 1b: per-faction masked column sums ----------------
__global__ void colsum_kernel(const float* __restrict__ newh, const int* __restrict__ positions,
                              float* __restrict__ fsum, float* __restrict__ fsum_p,
                              float* __restrict__ fsum_n) {
  int c = blockIdx.x * 256 + threadIdx.x;            // grid (4, 8): 1024 cols x 8 factions
  int f = blockIdx.y;
  float fs = 0.f, fp = 0.f, fn = 0.f;
  for (int r = f * 2048; r < (f + 1) * 2048; ++r) {
    float v = newh[(size_t)r * DH + c];
    int p = positions[r];
    fs += v;
    if (p > 0) fp += v; else if (p < 0) fn += v;
  }
  fsum  [f * DH + c] = fs;
  fsum_p[f * DH + c] = fp;
  fsum_n[f * DH + c] = fn;
}

// ---------------- sync stage 2: analytic means (group + post-sync faction + global) ----------------
__global__ void syncprep_kernel(const float* __restrict__ fsum, const float* __restrict__ fsum_p,
                                const float* __restrict__ fsum_n, const int* __restrict__ cnts,
                                float* __restrict__ mean1, float* __restrict__ mean2,
                                float* __restrict__ fmean, float* __restrict__ gmean) {
  int c = threadIdx.x;                               // 1 block, 1024 threads
  int c1 = cnts[0], c2 = cnts[1];
  float sp = 0.f, sn = 0.f;
  for (int f = 0; f < 8; ++f) { sp += fsum_p[f * DH + c]; sn += fsum_n[f * DH + c]; }
  float m1 = sp / (float)max(c1, 1);
  float m2 = sn / (float)max(c2, 1);
  mean1[c] = m1; mean2[c] = m2;
  float gacc = 0.f;
  for (int f = 0; f < 8; ++f) {
    float fpost = fsum[f * DH + c];
    if (c1 >= 2) fpost += 0.1f * ((float)cnts[2 + f]  * m1 - fsum_p[f * DH + c]);
    if (c2 >= 2) fpost += 0.1f * ((float)cnts[10 + f] * m2 - fsum_n[f * DH + c]);
    float fm = fpost * (1.f / 2048.f);
    fmean[f * DH + c] = fm;
    gacc += fm;
  }
  gmean[c] = gacc * 0.125f;
}

// ---------------- sync stage 3: elementwise blend (group sync + faction sync + debate) ----------------
__global__ void sync_apply_kernel(float* __restrict__ newh, const int* __restrict__ positions,
                                  const int* __restrict__ cnts,
                                  const float* __restrict__ mean1, const float* __restrict__ mean2,
                                  const float* __restrict__ fmean, const float* __restrict__ gmean,
                                  const int* __restrict__ step) {
  int c1 = cnts[0], c2 = cnts[1];
  int st = step[0];
#pragma unroll
  for (int i = 0; i < 4; ++i) {
    int idx = blockIdx.x * 1024 + i * 256 + threadIdx.x;   // grid 16384
    int r = idx >> 10;
    int c = idx & (DH - 1);
    float v = newh[idx];
    int p = positions[r];
    if (p > 0 && c1 >= 2)      v = 0.9f * v + 0.1f * mean1[c];
    else if (p < 0 && c2 >= 2) v = 0.9f * v + 0.1f * mean2[c];
    int f = r >> 11;
    v = 0.85f * v + 0.15f * fmean[f * DH + c];             // SYNC = 0.15
    if (st > 5 && (r & 2047) < 512)                        // debate head: fs/4 rows
      v = 0.85f * v + 0.15f * gmean[c];                    // DEBATE = 0.15
    newh[idx] = v;
  }
}

// ---------------- softmax(tension) @ output ----------------
__global__ void combine_kernel(const float* __restrict__ outF, const float* __restrict__ tension,
                               const float* __restrict__ scal, float* __restrict__ cvec) {
  int j = blockIdx.x * 256 + threadIdx.x;            // grid (2, 8)
  float m = scal[0], inv = 1.f / scal[1];
  int r0 = blockIdx.y * 2048;
  float acc = 0.f;
  for (int r = r0; r < r0 + 2048; ++r) {
    float w = expf(tension[r] - m) * inv;
    acc += w * outF[(size_t)r * DOUT + j];
  }
  atomicAdd(&cvec[j], acc);
}

// ---------------- pred = combined_out @ Wo + bo ----------------
__global__ void pred_kernel(const float* __restrict__ cvec, const float* __restrict__ Wo,
                            const float* __restrict__ bo, float* __restrict__ pred) {
  int j = blockIdx.x * 256 + threadIdx.x;            // grid 2
  float s = bo[j];
  for (int k = 0; k < DOUT; ++k) s += cvec[k] * Wo[(size_t)k * DIN + j];
  pred[j] = s;
}

// ==================================================================
extern "C" void kernel_launch(void* const* d_in, const int* in_sizes, int n_in,
                              void* d_out, int out_size, void* d_ws, size_t ws_size,
                              hipStream_t stream) {
  const float* x       = (const float*)d_in[0];
  const float* hiddens = (const float*)d_in[1];
  const float* wealth  = (const float*)d_in[2];
  const float* Wa1     = (const float*)d_in[3];
  const float* ba1     = (const float*)d_in[4];
  const float* Wa2     = (const float*)d_in[5];
  const float* ba2     = (const float*)d_in[6];
  const float* Wg1     = (const float*)d_in[7];
  const float* bg1     = (const float*)d_in[8];
  const float* Wg2     = (const float*)d_in[9];
  const float* bg2     = (const float*)d_in[10];
  const float* W_ih    = (const float*)d_in[11];
  const float* W_hh    = (const float*)d_in[12];
  const float* b_ih    = (const float*)d_in[13];
  const float* b_hh    = (const float*)d_in[14];
  const float* Wo      = (const float*)d_in[15];
  const float* bo      = (const float*)d_in[16];
  const int*   positions = (const int*)d_in[17];
  const int*   step      = (const int*)d_in[18];

  float* out  = (float*)d_out;
  float* pred = out;            // [512]
  float* avg  = out + 512;      // [1]
  float* newh = out + 513;      // [16384 x 1024]

  // ---- workspace layout (bytes, 256-aligned) ----
  char* w = (char*)d_ws;
  size_t off = 0;
  auto take = [&](size_t bytes) { size_t o = off; off = (off + bytes + 255) & ~(size_t)255; return o; };
  u16*   hBf     = (u16*)  (w + take((size_t)NC * DH * 2));        // 32 MB
  u16*   tAb     = (u16*)  (w + take((size_t)NC * 256 * 2));       //  8 MB
  u16*   outBf   = (u16*)  (w + take((size_t)NC * DOUT * 2));      // 16 MB
  float* outF    = (float*)(w + take((size_t)NC * DOUT * 4));      // 32 MB
  u16*   W1hT    = (u16*)  (w + take((size_t)256 * DH * 2));
  u16*   W2T     = (u16*)  (w + take((size_t)DOUT * 256 * 2));
  u16*   WihT    = (u16*)  (w + take((size_t)H3 * DIN * 2));
  u16*   WhhT    = (u16*)  (w + take((size_t)H3 * DH * 2));
  float* xabias  = (float*)(w + take(256 * 4));
  float* tacc    = (float*)(w + take((size_t)NC * 4));
  float* tension = (float*)(w + take((size_t)NC * 4));
  float* scal    = (float*)(w + take(256));
  float* fsum    = (float*)(w + take((size_t)8 * DH * 4));
  float* fsum_p  = (float*)(w + take((size_t)8 * DH * 4));
  float* fsum_n  = (float*)(w + take((size_t)8 * DH * 4));
  float* mean1   = (float*)(w + take((size_t)DH * 4));
  float* mean2   = (float*)(w + take((size_t)DH * 4));
  float* fmean   = (float*)(w + take((size_t)8 * DH * 4));
  float* gmean   = (float*)(w + take((size_t)DH * 4));
  int*   cnts    = (int*)  (w + take(128));
  float* cvec    = (float*)(w + take((size_t)DOUT * 4));

  // 0) zero accumulators
  init_kernel<<<64, 256, 0, stream>>>(tacc, cvec, cnts);

  // 1) bf16 conversions + K-major weight transposes
  conv_bf16_kernel<<<(NC * DH) / 1024, 256, 0, stream>>>(hBf, hiddens);
  transpose_conv_kernel<<<(DH * MW) / 256, 256, 0, stream>>>(W1hT, DH, Wa1 + (size_t)DIN * MW, MW, DH, MW, 1.f);
  transpose_conv_kernel<<<(DH * MW) / 256, 256, 0, stream>>>(W1hT + (size_t)MW * DH, DH, Wg1 + (size_t)DIN * MW, MW, DH, MW, 1.f);
  transpose_conv_kernel<<<(MW * DOUT) / 256, 256, 0, stream>>>(W2T, 256, Wa2, DOUT, MW, DOUT, 1.f);
  transpose_conv_kernel<<<(MW * DOUT) / 256, 256, 0, stream>>>(W2T + MW, 256, Wg2, DOUT, MW, DOUT, -1.f);
  transpose_conv_kernel<<<(DIN * H3) / 256, 256, 0, stream>>>(WihT, DIN, W_ih, H3, DIN, H3, 1.f);
  transpose_conv_kernel<<<(DH * H3) / 256, 256, 0, stream>>>(WhhT, DH, W_hh, H3, DH, H3, 1.f);

  // 2) fold x @ W1[:512] + b1 into per-column bias
  xbias_kernel<<<1, 256, 0, stream>>>(x, Wa1, ba1, Wg1, bg1, xabias);

  // 3) layer-1 MLP (both branches, N=256): t = relu(h @ W1h + xbias)
  gemm_bf16_kernel<0, DH><<<dim3(NC / 128, 2), 256, 0, stream>>>(
      hBf, W1hT, 256, tAb, nullptr, xabias, nullptr, nullptr);

  // 4) layer-2 fused a-g (K=256, N=512) + tension sum-of-squares
  gemm_bf16_kernel<1, 256><<<dim3(NC / 128, 4), 256, 0, stream>>>(
      tAb, W2T, DOUT, outBf, outF, ba2, bg2, tacc);

  // 5) tension, softmax stats, avg_tension
  tension_softmax_kernel<<<1, 256, 0, stream>>>(tacc, tension, scal, avg);

  // 6) fused GRU -> new_h (pre-sync) straight into d_out
  gru_kernel<<<dim3(NC / 64, DH / 64), 256, 0, stream>>>(
      outBf, hBf, WihT, WhhT, b_ih, b_hh, W_ih + (size_t)DIN * H3,
      tension, hiddens, wealth, newh);

  // 7) group / faction / debate sync (analytic means, two passes)
  counts_kernel<<<NC / 256, 256, 0, stream>>>(positions, cnts);
  colsum_kernel<<<dim3(4, 8), 256, 0, stream>>>(newh, positions, fsum, fsum_p, fsum_n);
  syncprep_kernel<<<1, 1024, 0, stream>>>(fsum, fsum_p, fsum_n, cnts, mean1, mean2, fmean, gmean);
  sync_apply_kernel<<<(NC * DH) / 1024, 256, 0, stream>>>(newh, positions, cnts, mean1, mean2, fmean, gmean, step);

  // 8) pred = softmax(tension) @ output @ Wo + bo
  combine_kernel<<<dim3(2, 8), 256, 0, stream>>>(outF, tension, scal, cvec);
  pred_kernel<<<2, 256, 0, stream>>>(cvec, Wo, bo, pred);
}
